// LitenBlock_84859963834929
// MI455X (gfx1250) — matrix operand order repaired and checked by reading, not compiled, not run on hardware
//
// CDNA5 (gfx1250) implementation of the Liten equivariant GNN forward pass.
// All dense contractions run through v_wmma_f32_16x16x32_f16 (wave32 WMMA,
// f32 accumulate); tiles are staged to LDS with async global->LDS loads
// (GLOBAL_LOAD_ASYNC_TO_LDS_B32, ASYNCcnt + s_wait_asynccnt). Each wave owns
// a 16x32 output tile (2 WMMAs per K-step, A-fragment reused). Gather/scatter
// (attention + segment_sum) uses VALU + global_atomic_add_f32.

#include <hip/hip_runtime.h>
#include <math.h>

#define Nn    8000
#define Cc    256
#define NRBFk 32
#define CUTV  5.0f

typedef __attribute__((ext_vector_type(16))) _Float16 v16h;
typedef __attribute__((ext_vector_type(8)))  float    v8f;

__device__ __forceinline__ float silu_f(float x) { return x / (1.0f + expf(-x)); }

// ---------------- async global -> LDS staging helpers ----------------
#if __has_builtin(__builtin_amdgcn_global_load_async_to_lds_b32)
#define HAVE_ASYNC_LDS 1
#endif

__device__ __forceinline__ void lds_copy_b32(const float* __restrict__ g, float* l) {
#ifdef HAVE_ASYNC_LDS
  __builtin_amdgcn_global_load_async_to_lds_b32(
      (__attribute__((address_space(1))) int*)g,
      (__attribute__((address_space(3))) int*)l, 0, 0);
#else
  *l = *g;
#endif
}

__device__ __forceinline__ void lds_copy_wait() {
#ifdef HAVE_ASYNC_LDS
#if __has_builtin(__builtin_amdgcn_s_wait_asynccnt)
  __builtin_amdgcn_s_wait_asynccnt(0);
#else
  asm volatile("s_wait_asynccnt 0" ::: "memory");
#endif
#endif
}

// ---------------- generic WMMA GEMM: Y = act(A[MxK] @ B[KxNc] + bias) ------
// Block: 256 threads = 8 waves. Block tile: 128(M) x 32(N). Each wave owns a
// 16x32 tile (two v_wmma_f32_16x16x32_f16 per K-step, A-fragment reused).
// Requires K % 32 == 0 and Nc % 32 == 0 (true for all call sites here).
// OOB M-rows: A addresses are clamped (not zeroed) -- a clamped row only
// feeds the matching OOB output row, which is masked at the store.
__global__ void __launch_bounds__(256)
gemm_wmma_kernel(const float* __restrict__ A, const float* __restrict__ B,
                 const float* __restrict__ bias, float* __restrict__ Y,
                 int M, int K, int Nc, int act)
{
  __shared__ float sA[128 * 32];   // 16 KB
  __shared__ float sB[32 * 32];    // 4 KB
  const int tid  = threadIdx.x;
  const int wave = tid >> 5;
  const int lane = tid & 31;
  const int half = lane >> 4;
  const int l15  = lane & 15;
  const long m0 = (long)blockIdx.x * 128;
  const long n0 = (long)blockIdx.y * 32;

  v8f acc0 = {};
  v8f acc1 = {};

  for (int k0 = 0; k0 < K; k0 += 32) {
    __syncthreads();  // previous iteration's LDS reads done before overwrite
    // ---- stage A tile (128 x 32 f32), clamped rows, no divergence ----
#pragma unroll
    for (int i = 0; i < 16; ++i) {
      int idx = tid + i * 256;
      int row = idx >> 5, col = idx & 31;
      long grow = m0 + row;
      grow = (grow < M) ? grow : 0;   // address select, not EXEC divergence
      lds_copy_b32(&A[grow * (long)K + k0 + col], &sA[idx]);
    }
    // ---- stage B tile (32 x 32 f32) ----
#pragma unroll
    for (int i = 0; i < 4; ++i) {
      int idx = tid + i * 256;
      int kk = idx >> 5, cc = idx & 31;
      lds_copy_b32(&B[(long)(k0 + kk) * Nc + n0 + cc], &sB[idx]);
    }
    // prefetch next K-slab of A (global_prefetch_b8)
    if (k0 + 32 < K) {
      long prow = m0 + (tid >> 1);
      prow = (prow < M) ? prow : 0;
      __builtin_prefetch(&A[prow * (long)K + k0 + 32], 0, 1);
    }
    lds_copy_wait();
    __syncthreads();

    // ---- build fragments in contiguous k order (same for A and B) ----
    v16h af = {};
#pragma unroll
    for (int e = 0; e < 16; ++e) {
      // A 16x32 f16: lane%16 = M, K = half*16 + elem
      int kk = half * 16 + e;
      af[e] = (_Float16)sA[(wave * 16 + l15) * 32 + kk];
    }
    v16h bf0 = {}, bf1 = {};
#pragma unroll
    for (int e = 0; e < 16; ++e) {
      // B 32x16 f16: lane%16 = N, K = half*16 + elem
      bf0[e] = (_Float16)sB[(half * 16 + e) * 32 + l15];
      bf1[e] = (_Float16)sB[(half * 16 + e) * 32 + 16 + l15];
    }
    acc0 = __builtin_amdgcn_wmma_f32_16x16x32_f16(
        false, af, false, bf0, (short)0, acc0, false, false);
    acc1 = __builtin_amdgcn_wmma_f32_16x16x32_f16(
        false, af, false, bf1, (short)0, acc1, false, false);
  }

  // ---- write out (D layout: M = r + 8*half, N = lane%16) ----
  const long col0 = n0 + l15;
  const long col1 = n0 + 16 + l15;
  float b0 = bias ? bias[col0] : 0.0f;
  float b1 = bias ? bias[col1] : 0.0f;
#pragma unroll
  for (int r = 0; r < 8; ++r) {
    long row = m0 + wave * 16 + half * 8 + r;
    if (row < M) {
      float v0 = acc0[r] + b0;
      float v1 = acc1[r] + b1;
      if (act) { v0 = silu_f(v0); v1 = silu_f(v1); }
      Y[row * (long)Nc + col0] = v0;
      Y[row * (long)Nc + col1] = v1;
    }
  }
}

// ---------------- edge geometry + RBF (folds cutoff into rbf) --------------
__global__ void edge_geom_kernel(const float* __restrict__ evec,
                                 float* __restrict__ ev, float* __restrict__ cut,
                                 float* __restrict__ rbf, int E)
{
  int e = blockIdx.x * blockDim.x + threadIdx.x;
  if (e >= E) return;
  float x = evec[e * 3 + 0], y = evec[e * 3 + 1], z = evec[e * 3 + 2];
  float d = sqrtf(x * x + y * y + z * z + 1e-16f);
  float inv = 1.0f / (d + 1e-8f);
  ev[e * 3 + 0] = x * inv; ev[e * 3 + 1] = y * inv; ev[e * 3 + 2] = z * inv;
  float cv = (d < CUTV) ? 0.5f * (cosf(3.14159265358979f * d / CUTV) + 1.0f) : 0.0f;
  cut[e] = cv;
  float start = expf(-CUTV);
  float bw = (2.0f / (float)NRBFk) * (1.0f - start);
  float beta = 1.0f / (bw * bw);
  float ed = expf(-(5.0f / CUTV) * d);
#pragma unroll
  for (int k = 0; k < NRBFk; ++k) {
    float mean = start + (1.0f - start) * ((float)k / (float)(NRBFk - 1));
    float t = ed - mean;
    rbf[(long)e * NRBFk + k] = expf(-beta * t * t) * cv;
  }
}

// ---------------- species embedding ----------------------------------------
__global__ void __launch_bounds__(256)
embed_kernel(const float* __restrict__ emb, const int* __restrict__ sp,
             float* __restrict__ nf)
{
  int n = blockIdx.x, c = threadIdx.x;
  nf[(long)n * Cc + c] = emb[sp[n] * Cc + c];
}

// ---------------- layernorm over C=256 (block per node) --------------------
__global__ void __launch_bounds__(256)
layernorm_kernel(const float* __restrict__ x, const float* __restrict__ g,
                 const float* __restrict__ b, float* __restrict__ y)
{
  __shared__ float s1[8], s2[8];
  int n = blockIdx.x, c = threadIdx.x;
  float v = x[(long)n * Cc + c];
  float a = v, q = v * v;
#pragma unroll
  for (int m = 16; m >= 1; m >>= 1) { a += __shfl_xor(a, m, 32); q += __shfl_xor(q, m, 32); }
  int wave = c >> 5, lane = c & 31;
  if (lane == 0) { s1[wave] = a; s2[wave] = q; }
  __syncthreads();
  float ta = 0.f, tq = 0.f;
#pragma unroll
  for (int j = 0; j < 8; ++j) { ta += s1[j]; tq += s2[j]; }
  float mean = ta / (float)Cc;
  float var  = tq / (float)Cc - mean * mean;
  y[(long)n * Cc + c] = (v - mean) * rsqrtf(var + 1e-5f) * g[c] + b[c];
}

// ---------------- attention + message (head dim 32 == one wave32) ----------
__global__ void __launch_bounds__(256)
attn_msg_kernel(const float* __restrict__ n_h, const float* __restrict__ e_h,
                const float* __restrict__ alpha, const float* __restrict__ cut,
                const int* __restrict__ snd, const int* __restrict__ rcv,
                float* __restrict__ msg)
{
  int e = blockIdx.x, c = threadIdx.x;
  int s = snd[e], r = rcv[e];
  float nhs = n_h[(long)s * Cc + c];
  float nhr = n_h[(long)r * Cc + c];
  float eh  = e_h[(long)e * Cc + c];
  float t = silu_f(nhr + nhs + eh) * alpha[c];
#pragma unroll
  for (int m = 16; m >= 1; m >>= 1) t += __shfl_xor(t, m, 32);  // sum over HD=32
  float a = t * cut[e];
  msg[(long)e * Cc + c] = nhs * eh * a;
}

// ---------------- message scatter (segment_sum via atomics) ----------------
__global__ void __launch_bounds__(256)
scatter_kernel(const float* __restrict__ msg, const float* __restrict__ nsc,
               const float* __restrict__ ev, const float* __restrict__ vf,
               const int* __restrict__ snd, const int* __restrict__ rcv,
               float* __restrict__ nacc, float* __restrict__ vacc, int first)
{
  int e = blockIdx.x, c = threadIdx.x;
  int r = rcv[e];
  atomicAdd(&nacc[(long)r * Cc + c], msg[(long)e * Cc + c]);
  float e0 = ev[e * 3 + 0], e1 = ev[e * 3 + 1], e2 = ev[e * 3 + 2];
  float v0, v1, v2;
  if (first) {
    float s1 = nsc[(long)e * Cc + c];
    v0 = s1 * e0; v1 = s1 * e1; v2 = s1 * e2;
  } else {
    int s = snd[e];
    float s1 = nsc[(long)e * 2 * Cc + c];
    float s2 = nsc[(long)e * 2 * Cc + Cc + c];
    v0 = vf[((long)s * 3 + 0) * Cc + c] * s1 + s2 * e0;
    v1 = vf[((long)s * 3 + 1) * Cc + c] * s1 + s2 * e1;
    v2 = vf[((long)s * 3 + 2) * Cc + c] * s1 + s2 * e2;
  }
  atomicAdd(&vacc[((long)r * 3 + 0) * Cc + c], v0);
  atomicAdd(&vacc[((long)r * 3 + 1) * Cc + c], v1);
  atomicAdd(&vacc[((long)r * 3 + 2) * Cc + c], v2);
}

// ---------------- cross-product edge feature (middle layers) ---------------
__global__ void __launch_bounds__(256)
cross_phi_kernel(const float* __restrict__ vfc, const float* __restrict__ ev,
                 const int* __restrict__ snd, const int* __restrict__ rcv,
                 float* __restrict__ phi)
{
  int e = blockIdx.x, c = threadIdx.x;
  int s = snd[e], r = rcv[e];
  float e0 = ev[e * 3 + 0], e1 = ev[e * 3 + 1], e2 = ev[e * 3 + 2];
  float a0 = vfc[((long)s * 3 + 0) * Cc + c];
  float a1 = vfc[((long)s * 3 + 1) * Cc + c];
  float a2 = vfc[((long)s * 3 + 2) * Cc + c];
  float b0 = vfc[((long)r * 3 + 0) * Cc + c];
  float b1 = vfc[((long)r * 3 + 1) * Cc + c];
  float b2 = vfc[((long)r * 3 + 2) * Cc + c];
  float ci0 = a1 * e2 - a2 * e1, ci1 = a2 * e0 - a0 * e2, ci2 = a0 * e1 - a1 * e0;
  float cj0 = b1 * e2 - b2 * e1, cj1 = b2 * e0 - b0 * e2, cj2 = b0 * e1 - b1 * e0;
  phi[(long)e * Cc + c] = ci0 * cj0 + ci1 * cj1 + ci2 * cj2;
}

// ---------------- vec_tri / vec_qua ----------------------------------------
__global__ void __launch_bounds__(256)
tri_qua_kernel(const float* __restrict__ vl, float* __restrict__ tri,
               float* __restrict__ qua)
{
  int n = blockIdx.x, c = threadIdx.x;
  float t = 0.f, q = 0.f;
#pragma unroll
  for (int k = 0; k < 3; ++k) {
    float v1 = vl[((long)n * 3 + k) * 512 + c];
    float v2 = vl[((long)n * 3 + k) * 512 + 256 + c];
    t += v1 * v2; q += v2 * v2;
  }
  float s = sqrtf(q + 1e-16f);
  tri[(long)n * Cc + c] = t;
  qua[(long)n * Cc + c] = s * s * s;
}

// ---------------- per-layer node/vector combine ----------------------------
__global__ void __launch_bounds__(256)
combine_kernel(const float* __restrict__ nfu, const float* __restrict__ tri,
               const float* __restrict__ qua, const float* __restrict__ vl,
               float* __restrict__ nf, float* __restrict__ vf, int cols, int last)
{
  int n = blockIdx.x, c = threadIdx.x;
  float s1 = nfu[(long)n * cols + c];
  float s2 = nfu[(long)n * cols + Cc + c];
  float add = (qua[(long)n * Cc + c] + tri[(long)n * Cc + c]) * s1 + s2;
  if (!last) {
    float s3 = nfu[(long)n * cols + 2 * Cc + c];
#pragma unroll
    for (int k = 0; k < 3; ++k)
      vf[((long)n * 3 + k) * Cc + c] += vl[((long)n * 3 + k) * 512 + c] * s3;
  }
  nf[(long)n * Cc + c] += add;
}

// ---------------- final readout (128 cols -> scalar) -----------------------
__global__ void __launch_bounds__(128)
readout_kernel(const float* __restrict__ h1, const float* __restrict__ Wr2,
               const float* __restrict__ br2, float* __restrict__ out)
{
  __shared__ float red[4];
  int n = blockIdx.x, c = threadIdx.x;
  float t = h1[(long)n * 128 + c] * Wr2[c];
#pragma unroll
  for (int m = 16; m >= 1; m >>= 1) t += __shfl_xor(t, m, 32);
  int wave = c >> 5, lane = c & 31;
  if (lane == 0) red[wave] = t;
  __syncthreads();
  if (c == 0) out[n] = red[0] + red[1] + red[2] + red[3] + br2[0];
}

// ---------------- grid-stride utilities ------------------------------------
__global__ void zero_kernel(float* p, long n) {
  for (long i = (long)blockIdx.x * 256 + threadIdx.x; i < n; i += (long)gridDim.x * 256)
    p[i] = 0.0f;
}
__global__ void add_inplace_kernel(float* d, const float* s, long n) {
  for (long i = (long)blockIdx.x * 256 + threadIdx.x; i < n; i += (long)gridDim.x * 256)
    d[i] += s[i];
}
__global__ void scale_mod_kernel(float* p, const float* w, long n) {
  for (long i = (long)blockIdx.x * 256 + threadIdx.x; i < n; i += (long)gridDim.x * 256)
    p[i] *= w[(int)(i & (Cc - 1))];
}
__global__ void fma_inplace_kernel(float* d, const float* a, const float* b, long n) {
  for (long i = (long)blockIdx.x * 256 + threadIdx.x; i < n; i += (long)gridDim.x * 256)
    d[i] += a[i] * b[i];
}

// ===========================================================================
extern "C" void kernel_launch(void* const* d_in, const int* in_sizes, int n_in,
                              void* d_out, int out_size, void* d_ws, size_t ws_size,
                              hipStream_t stream) {
  (void)n_in; (void)out_size; (void)ws_size;

  int idx = 0;
  auto F = [&](void) -> const float* { return (const float*)d_in[idx++]; };
  const float* edge_vectors = F();
  const int*   node_species = (const int*)d_in[idx++];
  const int*   senders      = (const int*)d_in[idx++];
  const int*   receivers    = (const int*)d_in[idx++];
  const int N = in_sizes[1];
  const int E = in_sizes[2];

  const float *p_embed = F(), *p_We0 = F(), *p_be0 = F();
  const float *p_Wr1 = F(), *p_br1 = F(), *p_Wr2 = F(), *p_br2 = F();
  const float *p_on_g = F(), *p_on_b = F();

  struct LP { const float *ln_g,*ln_b,*vln_w,*alpha,*Wvec,*Wn,*bn,*We,*be,
                           *Wp1,*bp1,*Wp2,*bp2,*Wc,*Wf,*bf; };
  LP L[4];
  for (int i = 0; i < 4; ++i) {
    L[i].ln_g = F(); L[i].ln_b = F(); L[i].vln_w = F(); L[i].alpha = F();
    L[i].Wvec = F(); L[i].Wn = F();   L[i].bn = F();    L[i].We = F();
    L[i].be = F();   L[i].Wp1 = F();  L[i].bp1 = F();   L[i].Wp2 = F();
    L[i].bp2 = F();
    if (i == 1 || i == 2) { L[i].Wc = F(); L[i].Wf = F(); L[i].bf = F(); }
    else                  { L[i].Wc = nullptr; L[i].Wf = nullptr; L[i].bf = nullptr; }
  }

  // ---- workspace carve (fp32) ----
  float* w = (float*)d_ws;
  size_t off = 0;
  auto alloc = [&](size_t n) { float* p = w + off; off += n; return p; };
  float* t_ev   = alloc((size_t)3 * E);
  float* t_cut  = alloc((size_t)E);
  float* t_rbf  = alloc((size_t)E * NRBFk);
  float* ef     = alloc((size_t)E * Cc);
  float* nf     = alloc((size_t)N * Cc);
  float* vf     = alloc((size_t)3 * N * Cc);
  float* t_sca  = alloc((size_t)N * Cc);
  float* t_nh   = alloc((size_t)N * Cc);
  float* t_eh   = alloc((size_t)E * Cc);          // also reused for silu(ef@Wf+bf)
  float* t_msg  = alloc((size_t)E * Cc);
  float* t_nsc  = alloc((size_t)2 * E * Cc);
  float* t_nacc = alloc((size_t)N * Cc);
  float* t_vacc = alloc((size_t)3 * N * Cc);
  float* t_vfc  = alloc((size_t)3 * N * Cc);
  float* t_phi  = alloc((size_t)E * Cc);
  float* t_vl   = alloc((size_t)6 * N * Cc);
  float* t_nfu  = alloc((size_t)3 * N * Cc);
  float* t_tri  = alloc((size_t)N * Cc);
  float* t_qua  = alloc((size_t)N * Cc);
  float* t_h1   = alloc((size_t)N * 128);

  auto gemm = [&](const float* A, const float* B, const float* bias, float* Y,
                  int M, int K, int Nc, int act) {
    dim3 g((unsigned)((M + 127) / 128), (unsigned)(Nc / 32));
    gemm_wmma_kernel<<<g, 256, 0, stream>>>(A, B, bias, Y, M, K, Nc, act);
  };
  auto zero = [&](float* p, long n)               { zero_kernel<<<2048, 256, 0, stream>>>(p, n); };
  auto addi = [&](float* d, const float* s, long n){ add_inplace_kernel<<<2048, 256, 0, stream>>>(d, s, n); };

  // ---- edge geometry + ef = (rbf * cut) @ We0 + be0 ----
  edge_geom_kernel<<<(E + 255) / 256, 256, 0, stream>>>(edge_vectors, t_ev, t_cut, t_rbf, E);
  gemm(t_rbf, p_We0, p_be0, ef, E, NRBFk, Cc, 0);
  embed_kernel<<<N, Cc, 0, stream>>>(p_embed, node_species, nf);

  for (int i = 0; i < 4; ++i) {
    const bool first = (i == 0), last = (i == 3), mid = !first && !last;
    const LP& P = L[i];

    layernorm_kernel<<<N, Cc, 0, stream>>>(nf, P.ln_g, P.ln_b, t_sca);
    if (!first)
      scale_mod_kernel<<<2048, 256, 0, stream>>>(vf, P.vln_w, (long)3 * N * Cc);

    gemm(ef, P.We, P.be, t_eh, E, Cc, Cc, 1);            // e_h = silu(ef@We+be)
    gemm(t_sca, P.Wn, P.bn, t_nh, N, Cc, Cc, 0);         // n_h

    attn_msg_kernel<<<E, Cc, 0, stream>>>(t_nh, t_eh, P.alpha, t_cut,
                                          senders, receivers, t_msg);

    const int p1c = first ? Cc : 2 * Cc;
    gemm(t_msg, P.Wp1, P.bp1, t_nsc, E, Cc, p1c, 1);     // node_sca

    zero(t_nacc, (long)N * Cc);
    float* vtarget = first ? vf : t_vacc;
    zero(vtarget, (long)3 * N * Cc);
    scatter_kernel<<<E, Cc, 0, stream>>>(t_msg, t_nsc, t_ev, vf,
                                         senders, receivers, t_nacc, vtarget,
                                         first ? 1 : 0);

    if (mid) {
      gemm(vf, P.Wc, nullptr, t_vfc, 3 * N, Cc, Cc, 0);  // vfc = vf@Wc
      cross_phi_kernel<<<E, Cc, 0, stream>>>(t_vfc, t_ev, senders, receivers, t_phi);
      gemm(ef, P.Wf, P.bf, t_eh, E, Cc, Cc, 1);          // reuse t_eh
      fma_inplace_kernel<<<2048, 256, 0, stream>>>(ef, t_eh, t_phi, (long)E * Cc);
    }

    addi(nf, t_nacc, (long)N * Cc);                      // nf += scalar_out
    if (!first) addi(vf, t_vacc, (long)3 * N * Cc);      // vf += vector_out

    gemm(vf, P.Wvec, nullptr, t_vl, 3 * N, Cc, 2 * Cc, 0);
    tri_qua_kernel<<<N, Cc, 0, stream>>>(t_vl, t_tri, t_qua);

    const int p2c = last ? 2 * Cc : 3 * Cc;
    gemm(nf, P.Wp2, P.bp2, t_nfu, N, Cc, p2c, 0);
    combine_kernel<<<N, Cc, 0, stream>>>(t_nfu, t_tri, t_qua, t_vl, nf, vf,
                                         p2c, last ? 1 : 0);
  }

  // ---- readout ----
  layernorm_kernel<<<N, Cc, 0, stream>>>(nf, p_on_g, p_on_b, t_sca);
  gemm(t_sca, p_Wr1, p_br1, t_h1, N, Cc, 128, 1);
  readout_kernel<<<N, 128, 0, stream>>>(t_h1, p_Wr2, p_br2, (float*)d_out);
}
